// QLoRADense_89215060673243
// MI455X (gfx1250) — compile-verified, matrix operand order checked by reading, not run on hardware
//
#include <hip/hip_runtime.h>

// ---------------- types ----------------
typedef _Float16 h16 __attribute__((ext_vector_type(16)));
typedef _Float16 h8  __attribute__((ext_vector_type(8)));
typedef float    f8  __attribute__((ext_vector_type(8)));
typedef float    v8f __attribute__((ext_vector_type(8)));

#define D_IN        1024
#define FEAT        1024
#define RANK        16
#define ROWS_TOTAL  32768         // 4 * 8192
#define NTILES      (FEAT / 16)   // 64  (16-wide N tiles)
#define KSTEPS      (D_IN / 64)   // 16  (64-wide K iterations)

// LDS A-tile row stride in halves (64 data + 8 pad -> conflict-free, 16B aligned)
#define ASTRIDE 72

// ---------------------------------------------------------------------------
// Kernel 1: partial max|W| reduction. 256 blocks x 256 threads x 16 elements.
// ---------------------------------------------------------------------------
__global__ void k_amax_partial(const float* __restrict__ w, float* __restrict__ partial) {
    __shared__ float red[256];
    const int t = threadIdx.x;
    const int base = blockIdx.x * (256 * 16);
    float m = 0.0f;
#pragma unroll
    for (int i = 0; i < 16; ++i)
        m = fmaxf(m, fabsf(w[base + i * 256 + t]));
    red[t] = m;
    __syncthreads();
#pragma unroll
    for (int s = 128; s > 0; s >>= 1) {
        if (t < s) red[t] = fmaxf(red[t], red[t + s]);
        __syncthreads();
    }
    if (t == 0) partial[blockIdx.x] = red[0];
}

// ---------------------------------------------------------------------------
// Kernel 2: reduce 256 partials -> scale = amax / 7
// ---------------------------------------------------------------------------
__global__ void k_scale(const float* __restrict__ partial, float* __restrict__ scale) {
    __shared__ float red[256];
    const int t = threadIdx.x;
    red[t] = partial[t];
    __syncthreads();
#pragma unroll
    for (int s = 128; s > 0; s >>= 1) {
        if (t < s) red[t] = fmaxf(red[t], red[t + s]);
        __syncthreads();
    }
    if (t == 0) *scale = red[0] / 7.0f;
}

// ---------------------------------------------------------------------------
// Kernel 3: build W' = dequant(quant(W)) + a@b in f16, pre-packed into the
// WMMA B-matrix (32x16 f16, K-major-per-lane) fragment layout:
//   fragment id  = kTile (k/32) * NTILES + nTile (n/16)
//   lane         = (n%16) + 16*((k%16)/8)
//   element j    : j<8  -> K = (lane/16)*8 + j
//                  j>=8 -> K = 16 + (lane/16)*8 + (j-8)
// so each lane's 16 halves (32B) are contiguous in memory.
// ---------------------------------------------------------------------------
__global__ void k_pack(const float* __restrict__ w, const float* __restrict__ a,
                       const float* __restrict__ b, const float* __restrict__ scalep,
                       _Float16* __restrict__ packed) {
    const int idx = blockIdx.x * 256 + threadIdx.x;   // d*1024 + f
    const int d = idx >> 10;    // K index
    const int f = idx & 1023;   // N index
    const float s = *scalep;
    float q = rintf(w[idx] / s);                      // round-half-even like jnp.round
    q = fminf(fmaxf(q, -8.0f), 7.0f);
    float val = q * s;
    float lora = 0.0f;
#pragma unroll
    for (int r = 0; r < RANK; ++r)
        lora += a[d * RANK + r] * b[r * FEAT + f];
    val += lora;

    const int kTile = d >> 5;
    const int kl    = d & 31;
    const int nTile = f >> 4;
    const int nl    = f & 15;
    const int hi    = kl >> 4;           // 0: K<16, 1: K>=16
    const int rem   = kl & 15;
    const int khalf = rem >> 3;          // lane group select
    const int j     = (rem & 7) + hi * 8;
    const int lane  = nl + 16 * khalf;
    packed[((((size_t)kTile * NTILES) + nTile) * 32 + lane) * 16 + j] = (_Float16)val;
}

// ---------------------------------------------------------------------------
// Kernel 4: main GEMM  out[M,N] = x[M,K] @ W'[K,N] + bias
// Block: 256 threads (8 waves), tile 128(M) x 128(N).
// Wave grid 2(M) x 4(N); each wave: 4x2 tiles of 16x16, K unrolled by 64
// -> 16 WMMA per barrier. Double-buffered LDS for the fp32->fp16 x slab.
// ---------------------------------------------------------------------------
__global__ __launch_bounds__(256)
void k_gemm(const float* __restrict__ x, const _Float16* __restrict__ packedB,
            const float* __restrict__ bias, float* __restrict__ out) {
    // two ping-pong A tiles: 128 rows x 64 k halves, stride 72 (36 KB total)
    __shared__ _Float16 aTile[2][128 * ASTRIDE];

    const int t     = threadIdx.x;
    const int lane  = t & 31;
    const int wave  = t >> 5;
    const int wm    = wave & 1;    // M half within block tile
    const int wn    = wave >> 1;   // N quarter within block tile
    const int blockM = blockIdx.x * 128;
    const int blockN = blockIdx.y * 128;

    const int lrow  = t >> 1;      // 0..127: fill row
    const int lhalf = t & 1;       // 0..1: which 32-float half of the 64-slab
    const int khgrp = lane >> 4;   // lane/16
    const int lmod  = lane & 15;

    v8f acc[4][2];
#pragma unroll
    for (int i = 0; i < 4; ++i)
#pragma unroll
        for (int j = 0; j < 2; ++j)
            acc[i][j] = (v8f)0.0f;

    const float* xrow = x + (size_t)(blockM + lrow) * D_IN + lhalf * 32;

    // ---- prologue: fill buffer 0 with slab 0 ----
    {
        f8 a0 = *(const f8*)(xrow);
        f8 a1 = *(const f8*)(xrow + 8);
        f8 a2 = *(const f8*)(xrow + 16);
        f8 a3 = *(const f8*)(xrow + 24);
        _Float16* d = &aTile[0][lrow * ASTRIDE + lhalf * 32];
        *(h8*)(d)      = __builtin_convertvector(a0, h8);
        *(h8*)(d + 8)  = __builtin_convertvector(a1, h8);
        *(h8*)(d + 16) = __builtin_convertvector(a2, h8);
        *(h8*)(d + 24) = __builtin_convertvector(a3, h8);
    }
    __syncthreads();

    for (int kk = 0; kk < KSTEPS; ++kk) {
        const int cur = kk & 1;
        // ---- issue next slab's global loads early (clamped: last iter re-reads) ----
        const int nxt = (kk + 1 < KSTEPS) ? (kk + 1) : (KSTEPS - 1);
        const float* nsrc = xrow + nxt * 64;
        f8 n0 = *(const f8*)(nsrc);
        f8 n1 = *(const f8*)(nsrc + 8);
        f8 n2 = *(const f8*)(nsrc + 16);
        f8 n3 = *(const f8*)(nsrc + 24);
        __builtin_prefetch(xrow + (kk + 2) * 64, 0, 1);   // speculative, OOB-safe

        // ---- B fragments for both 32-wide k-substeps (L2-resident, pre-packed) ----
        h16 bfr[2][2];
#pragma unroll
        for (int s = 0; s < 2; ++s) {
#pragma unroll
            for (int nt = 0; nt < 2; ++nt) {
                const int kTile = kk * 2 + s;
                const int nTile = (blockN >> 4) + wn * 2 + nt;
                bfr[s][nt] = *(const h16*)(packedB +
                    ((((size_t)kTile * NTILES) + nTile) * 32 + lane) * 16);
            }
        }

        // ---- 16 WMMAs against the current LDS buffer ----
#pragma unroll
        for (int s = 0; s < 2; ++s) {
            h16 afr[4];
#pragma unroll
            for (int mt = 0; mt < 4; ++mt) {
                const _Float16* ap =
                    &aTile[cur][(wm * 64 + mt * 16 + lmod) * ASTRIDE + s * 32 + khgrp * 8];
                h8 a0 = *(const h8*)(ap);
                h8 a1 = *(const h8*)(ap + 16);
                afr[mt] = __builtin_shufflevector(a0, a1, 0, 1, 2, 3, 4, 5, 6, 7,
                                                  8, 9, 10, 11, 12, 13, 14, 15);
            }
#pragma unroll
            for (int nt = 0; nt < 2; ++nt) {
#pragma unroll
                for (int mt = 0; mt < 4; ++mt) {
                    acc[mt][nt] = __builtin_amdgcn_wmma_f32_16x16x32_f16(
                        false, afr[mt], false, bfr[s][nt], (short)0, acc[mt][nt],
                        false, false);
                }
            }
        }

        // ---- convert + store next slab into the other buffer ----
        {
            _Float16* d = &aTile[1 - cur][lrow * ASTRIDE + lhalf * 32];
            *(h8*)(d)      = __builtin_convertvector(n0, h8);
            *(h8*)(d + 8)  = __builtin_convertvector(n1, h8);
            *(h8*)(d + 16) = __builtin_convertvector(n2, h8);
            *(h8*)(d + 24) = __builtin_convertvector(n3, h8);
        }
        __syncthreads();   // single barrier per 64-wide k iteration
    }

    // ---- epilogue: C layout lane=(n%16)+16*(m/8), vgpr v -> m = v + 8*(lane/16) ----
#pragma unroll
    for (int nt = 0; nt < 2; ++nt) {
        const int col = blockN + wn * 32 + nt * 16 + lmod;
        const float bv = bias[col];
#pragma unroll
        for (int mt = 0; mt < 4; ++mt) {
            const int rbase = blockM + wm * 64 + mt * 16 + khgrp * 8;
#pragma unroll
            for (int v = 0; v < 8; ++v) {
                out[(size_t)(rbase + v) * FEAT + col] = acc[mt][nt][v] + bv;
            }
        }
    }
}

// ---------------------------------------------------------------------------
extern "C" void kernel_launch(void* const* d_in, const int* in_sizes, int n_in,
                              void* d_out, int out_size, void* d_ws, size_t ws_size,
                              hipStream_t stream) {
    const float* x      = (const float*)d_in[0];
    const float* kernel = (const float*)d_in[1];
    const float* bias   = (const float*)d_in[2];
    const float* a      = (const float*)d_in[3];
    const float* b      = (const float*)d_in[4];
    float* out = (float*)d_out;

    char* ws = (char*)d_ws;
    _Float16* packed = (_Float16*)ws;                        // 2 MiB packed W'
    float* scale     = (float*)(ws + (2u << 20));            // 4 B
    float* partial   = (float*)(ws + (2u << 20) + 256);      // 1 KiB partial maxima

    k_amax_partial<<<256, 256, 0, stream>>>(kernel, partial);
    k_scale<<<1, 256, 0, stream>>>(partial, scale);
    k_pack<<<(D_IN * FEAT) / 256, 256, 0, stream>>>(kernel, a, b, scale, packed);

    dim3 grid(ROWS_TOTAL / 128, FEAT / 128);
    k_gemm<<<grid, 256, 0, stream>>>(x, packed, bias, out);
}
